// GNNPolicy_87230785782197
// MI455X (gfx1250) — matrix operand order repaired and checked by reading, not compiled, not run on hardware
//
#include <hip/hip_runtime.h>

#define NCONS 100000
#define NVARS 200000
#define NEDGE 2000000

typedef __attribute__((ext_vector_type(16))) _Float16 v16h;
typedef __attribute__((ext_vector_type(8)))  _Float16 v8h;
typedef __attribute__((ext_vector_type(8)))  float    v8f;

__device__ __forceinline__ v8f wmma16(v16h a, v16h b, v8f c) {
  return __builtin_amdgcn_wmma_f32_16x16x32_f16(false, a, false, b, (short)0, c, false, false);
}
__device__ __forceinline__ v8f zero8() {
  v8f z = {0.f,0.f,0.f,0.f,0.f,0.f,0.f,0.f}; return z;
}

// B fragment: 32(K) x 16(N) tile from row-major f16 weight [rows x 64].
__device__ __forceinline__ v16h load_B(const _Float16* __restrict__ W, int k0, int n0, int lane) {
  int col = n0 + (lane & 15);
  int kb  = k0 + ((lane & 16) >> 1);
  v16h b;
#pragma unroll
  for (int h = 0; h < 16; ++h) b[h] = W[(kb + (h < 8 ? h : h + 8)) * 64 + col];
  return b;
}

// A fragment (16M x 32K) from f32 row (lane's row = lane&15 of the tile).
__device__ __forceinline__ v16h load_A_f32(const float* __restrict__ Xrow, int k0, int lane) {
  int kb = k0 + ((lane & 16) >> 1);
  v16h a;
#pragma unroll
  for (int i = 0; i < 8; ++i) {
    a[i]     = (_Float16)Xrow[kb + i];
    a[i + 8] = (_Float16)Xrow[kb + 16 + i];
  }
  return a;
}

__device__ __forceinline__ v16h load_A_f16(const _Float16* __restrict__ Xrow, int k0, int lane) {
  int kb = k0 + ((lane & 16) >> 1);
  v16h a;
#pragma unroll
  for (int i = 0; i < 8; ++i) {
    a[i]     = Xrow[kb + i];
    a[i + 8] = Xrow[kb + 16 + i];
  }
  return a;
}

// ---------------- weight conversion ----------------
__global__ void k_cvt(const float* __restrict__ s, _Float16* __restrict__ d, int n) {
  int i = blockIdx.x * blockDim.x + threadIdx.x;
  if (i < n) d[i] = (_Float16)s[i];
}
__global__ void k_cvt_pad(const float* __restrict__ s, _Float16* __restrict__ d, int validRows) {
  int i = blockIdx.x * blockDim.x + threadIdx.x;
  if (i < 32 * 64) {
    int r = i >> 6;
    d[i] = (r < validRows) ? (_Float16)s[i] : (_Float16)0.0f;
  }
}

// ---------------- node embedding: relu(relu(X@W0+b0)@W1+b1) ----------------
template <int INF>
__global__ void __launch_bounds__(256)
k_embed(const float* __restrict__ X,
        const _Float16* __restrict__ W0 /*32x64*/, const float* __restrict__ b0,
        const _Float16* __restrict__ W1 /*64x64*/, const float* __restrict__ b1,
        float* __restrict__ Y, int n)
{
  __shared__ _Float16 lsh[8][16 * 64];
  int lane  = threadIdx.x & 31;
  int wslot = threadIdx.x >> 5;
  int wave  = blockIdx.x * (blockDim.x >> 5) + wslot;
  int nWaves = gridDim.x * (blockDim.x >> 5);
  int nTiles = (n + 15) >> 4;
  int col = lane & 15;
  int hh  = (lane & 16) >> 1;
  int kbA = (lane & 16) >> 1;

  v16h B0[4];
#pragma unroll
  for (int nt = 0; nt < 4; ++nt) B0[nt] = load_B(W0, 0, nt * 16, lane);
  v16h B1a[4], B1b[4];
#pragma unroll
  for (int nt = 0; nt < 4; ++nt) { B1a[nt] = load_B(W1, 0, nt * 16, lane); B1b[nt] = load_B(W1, 32, nt * 16, lane); }

  float b0l[4], b1l[4];
#pragma unroll
  for (int nt = 0; nt < 4; ++nt) { b0l[nt] = b0[nt * 16 + col]; b1l[nt] = b1[nt * 16 + col]; }

  _Float16* lb = &lsh[wslot][hh * 64 + col];       // tile-LDS base (constant offsets below)

  for (int t = wave; t < nTiles; t += nWaves) {
    int row  = t * 16 + col;
    int rowc = row < n ? row : n - 1;
    bool full = (t * 16 + 16 <= n);                // wave-uniform
    const float* xr = X + (size_t)rowc * INF;
    v16h A;
#pragma unroll
    for (int i = 0; i < 8; ++i) {
      int k1 = kbA + i, k2 = kbA + 16 + i;         // folds vs compile-time INF
      A[i]     = (k1 < INF) ? (_Float16)xr[k1] : (_Float16)0.0f;
      A[i + 8] = (k2 < INF) ? (_Float16)xr[k2] : (_Float16)0.0f;
    }
#pragma unroll
    for (int nt = 0; nt < 4; ++nt) {
      v8f c = zero8();
      c = wmma16(A, B0[nt], c);
#pragma unroll
      for (int r = 0; r < 8; ++r) {
        float h = c[r] + b0l[nt];
        h = h > 0.f ? h : 0.f;
        lb[r * 64 + nt * 16] = (_Float16)h;
      }
    }
    asm volatile("" ::: "memory");
    v16h H0 = load_A_f16(&lsh[wslot][(lane & 15) * 64], 0, lane);
    v16h H1 = load_A_f16(&lsh[wslot][(lane & 15) * 64], 32, lane);
    asm volatile("" ::: "memory");
    v8f cc[4];
#pragma unroll
    for (int nt = 0; nt < 4; ++nt) {
      v8f c = zero8();
      c = wmma16(H0, B1a[nt], c);
      c = wmma16(H1, B1b[nt], c);
      cc[nt] = c;
    }
    float* yb = Y + (size_t)(t * 16 + hh) * 64 + col;
    if (full) {
#pragma unroll
      for (int nt = 0; nt < 4; ++nt)
#pragma unroll
        for (int r = 0; r < 8; ++r) {
          float h = cc[nt][r] + b1l[nt];
          yb[r * 64 + nt * 16] = h > 0.f ? h : 0.f;
        }
    } else {
#pragma unroll
      for (int nt = 0; nt < 4; ++nt)
#pragma unroll
        for (int r = 0; r < 8; ++r) {
          int orow = t * 16 + r + hh;
          if (orow < n) {
            float h = cc[nt][r] + b1l[nt];
            yb[r * 64 + nt * 16] = h > 0.f ? h : 0.f;
          }
        }
    }
  }
}

// ---------------- per-node linear precompute: Y = f16(X@W (+bias)) ----------------
__global__ void __launch_bounds__(256)
k_node_lin(const float* __restrict__ X, const _Float16* __restrict__ W,
           const float* __restrict__ bias, _Float16* __restrict__ Y, int n)
{
  int lane = threadIdx.x & 31;
  int wave = blockIdx.x * (blockDim.x >> 5) + (threadIdx.x >> 5);
  int nWaves = gridDim.x * (blockDim.x >> 5);
  int nTiles = (n + 15) >> 4;
  int col = lane & 15;
  int hh  = (lane & 16) >> 1;

  v16h Ba[4], Bb[4];
#pragma unroll
  for (int nt = 0; nt < 4; ++nt) { Ba[nt] = load_B(W, 0, nt * 16, lane); Bb[nt] = load_B(W, 32, nt * 16, lane); }
  float bl[4];
#pragma unroll
  for (int nt = 0; nt < 4; ++nt) bl[nt] = bias ? bias[nt * 16 + col] : 0.f;

  for (int t = wave; t < nTiles; t += nWaves) {
    int row  = t * 16 + col;
    int rowc = row < n ? row : n - 1;
    bool full = (t * 16 + 16 <= n);
    const float* xr = X + (size_t)rowc * 64;
    v16h A0 = load_A_f32(xr, 0, lane);
    v16h A1 = load_A_f32(xr, 32, lane);
    v8f cc[4];
#pragma unroll
    for (int nt = 0; nt < 4; ++nt) {
      v8f c = zero8();
      c = wmma16(A0, Ba[nt], c);
      c = wmma16(A1, Bb[nt], c);
      cc[nt] = c;
    }
    _Float16* yb = Y + (size_t)(t * 16 + hh) * 64 + col;
    if (full) {
#pragma unroll
      for (int nt = 0; nt < 4; ++nt)
#pragma unroll
        for (int r = 0; r < 8; ++r)
          yb[r * 64 + nt * 16] = (_Float16)(cc[nt][r] + bl[nt]);
    } else {
#pragma unroll
      for (int nt = 0; nt < 4; ++nt)
#pragma unroll
        for (int r = 0; r < 8; ++r) {
          int orow = t * 16 + r + hh;
          if (orow < n) yb[r * 64 + nt * 16] = (_Float16)(cc[nt][r] + bl[nt]);
        }
    }
  }
}

// ---------------- edge kernel: msg = relu(R[dst]+L[src]+ef*ew) @ fw + fb ; agg[dst]+=msg ----------------
__global__ void __launch_bounds__(256)
k_edge(const int* __restrict__ dstIdx, const int* __restrict__ srcIdx,
       const float* __restrict__ ef, const _Float16* __restrict__ Rh,
       const _Float16* __restrict__ Lh, const _Float16* __restrict__ ewh,
       const _Float16* __restrict__ fwh, const float* __restrict__ fb,
       float* __restrict__ agg, int nE)
{
  int lane = threadIdx.x & 31;
  int wave = blockIdx.x * (blockDim.x >> 5) + (threadIdx.x >> 5);
  int nWaves = gridDim.x * (blockDim.x >> 5);
  int nTiles = (nE + 15) >> 4;
  int col = lane & 15;
  int hh  = (lane & 16) >> 1;
  int kb  = (lane & 16) >> 1;

  v16h Ba[4], Bb[4];
#pragma unroll
  for (int nt = 0; nt < 4; ++nt) { Ba[nt] = load_B(fwh, 0, nt * 16, lane); Bb[nt] = load_B(fwh, 32, nt * 16, lane); }

  _Float16 ewa[2][16];
#pragma unroll
  for (int kt = 0; kt < 2; ++kt)
#pragma unroll
    for (int i = 0; i < 8; ++i) {
      ewa[kt][i]     = ewh[kt * 32 + kb + i];
      ewa[kt][8 + i] = ewh[kt * 32 + kb + 16 + i];
    }
  float fbl[4];
#pragma unroll
  for (int nt = 0; nt < 4; ++nt) fbl[nt] = fb[nt * 16 + col];

  for (int t = wave; t < nTiles; t += nWaves) {
    int row  = t * 16 + col;
    int rowc = row < nE ? row : nE - 1;
    bool full = (t * 16 + 16 <= nE);
    int d = dstIdx[rowc];
    int s = srcIdx[rowc];
    _Float16 w = (_Float16)(row < nE ? ef[rowc] : 0.f);
    const _Float16* rp = Rh + (size_t)d * 64;
    const _Float16* lp = Lh + (size_t)s * 64;

    v16h A[2];
#pragma unroll
    for (int kt = 0; kt < 2; ++kt) {
      int off = kt * 32 + kb;
      v8h r0 = *(const v8h*)(rp + off);
      v8h r1 = *(const v8h*)(rp + off + 16);
      v8h l0 = *(const v8h*)(lp + off);
      v8h l1 = *(const v8h*)(lp + off + 16);
#pragma unroll
      for (int i = 0; i < 8; ++i) {
        _Float16 x = (_Float16)(r0[i] + l0[i] + w * ewa[kt][i]);
        _Float16 y = (_Float16)(r1[i] + l1[i] + w * ewa[kt][8 + i]);
        A[kt][i]     = x > (_Float16)0.0f ? x : (_Float16)0.0f;
        A[kt][i + 8] = y > (_Float16)0.0f ? y : (_Float16)0.0f;
      }
    }
    v8f cc[4];
#pragma unroll
    for (int nt = 0; nt < 4; ++nt) {
      v8f c = zero8();
      c = wmma16(A[0], Ba[nt], c);
      c = wmma16(A[1], Bb[nt], c);
      cc[nt] = c;
    }
    if (full) {
#pragma unroll
      for (int r = 0; r < 8; ++r) {
        int dn = __shfl(d, r + hh, 32);
        float* ab = agg + (size_t)dn * 64 + col;
#pragma unroll
        for (int nt = 0; nt < 4; ++nt)
          atomicAdd(&ab[nt * 16], cc[nt][r] + fbl[nt]);
      }
    } else {
#pragma unroll
      for (int r = 0; r < 8; ++r) {
        int er = t * 16 + r + hh;
        int dn = __shfl(d, r + hh, 32);
        float* ab = agg + (size_t)dn * 64 + col;
#pragma unroll
        for (int nt = 0; nt < 4; ++nt)
          if (er < nE) atomicAdd(&ab[nt * 16], cc[nt][r] + fbl[nt]);
      }
    }
  }
}

// ---------------- conv output: Y = relu(concat(agg,right)@W0+b0) @ W1 + b1 ----------------
__global__ void __launch_bounds__(256)
k_update(const float* __restrict__ agg, const float* __restrict__ right,
         const _Float16* __restrict__ W0 /*128x64*/, const float* __restrict__ b0,
         const _Float16* __restrict__ W1 /*64x64*/, const float* __restrict__ b1,
         float* __restrict__ Y, int n)
{
  __shared__ _Float16 lsh[8][16 * 64];
  int lane  = threadIdx.x & 31;
  int wslot = threadIdx.x >> 5;
  int wave  = blockIdx.x * (blockDim.x >> 5) + wslot;
  int nWaves = gridDim.x * (blockDim.x >> 5);
  int nTiles = (n + 15) >> 4;
  int col = lane & 15;
  int hh  = (lane & 16) >> 1;

  v16h B0[4][4];
#pragma unroll
  for (int nt = 0; nt < 4; ++nt)
#pragma unroll
    for (int kt = 0; kt < 4; ++kt) B0[nt][kt] = load_B(W0, kt * 32, nt * 16, lane);
  float b0l[4], b1l[4];
#pragma unroll
  for (int nt = 0; nt < 4; ++nt) { b0l[nt] = b0[nt * 16 + col]; b1l[nt] = b1[nt * 16 + col]; }

  _Float16* lb = &lsh[wslot][hh * 64 + col];

  for (int t = wave; t < nTiles; t += nWaves) {
    int row  = t * 16 + col;
    int rowc = row < n ? row : n - 1;
    bool full = (t * 16 + 16 <= n);
    const float* ar = agg + (size_t)rowc * 64;
    const float* rr = right + (size_t)rowc * 64;
    v16h A[4];
    A[0] = load_A_f32(ar, 0, lane);
    A[1] = load_A_f32(ar, 32, lane);
    A[2] = load_A_f32(rr, 0, lane);
    A[3] = load_A_f32(rr, 32, lane);
#pragma unroll
    for (int nt = 0; nt < 4; ++nt) {
      v8f c = zero8();
#pragma unroll
      for (int kt = 0; kt < 4; ++kt) c = wmma16(A[kt], B0[nt][kt], c);
#pragma unroll
      for (int r = 0; r < 8; ++r) {
        float h = c[r] + b0l[nt];
        h = h > 0.f ? h : 0.f;
        lb[r * 64 + nt * 16] = (_Float16)h;
      }
    }
    asm volatile("" ::: "memory");
    v16h H0 = load_A_f16(&lsh[wslot][(lane & 15) * 64], 0, lane);
    v16h H1 = load_A_f16(&lsh[wslot][(lane & 15) * 64], 32, lane);
    asm volatile("" ::: "memory");
    v8f cc[4];
#pragma unroll
    for (int nt = 0; nt < 4; ++nt) {
      v8f c = zero8();
      c = wmma16(H0, load_B(W1, 0, nt * 16, lane), c);
      c = wmma16(H1, load_B(W1, 32, nt * 16, lane), c);
      cc[nt] = c;
    }
    float* yb = Y + (size_t)(t * 16 + hh) * 64 + col;
    if (full) {
#pragma unroll
      for (int nt = 0; nt < 4; ++nt)
#pragma unroll
        for (int r = 0; r < 8; ++r)
          yb[r * 64 + nt * 16] = cc[nt][r] + b1l[nt];
    } else {
#pragma unroll
      for (int nt = 0; nt < 4; ++nt)
#pragma unroll
        for (int r = 0; r < 8; ++r) {
          int orow = t * 16 + r + hh;
          if (orow < n) yb[r * 64 + nt * 16] = cc[nt][r] + b1l[nt];
        }
    }
  }
}

// ---------------- head: out = relu(X@W0+b0) . w1 (+ b1) ----------------
__global__ void __launch_bounds__(256)
k_head(const float* __restrict__ X, const _Float16* __restrict__ W0,
       const float* __restrict__ b0, const float* __restrict__ w1,
       const float* __restrict__ b1, float* __restrict__ out, int n)
{
  int lane = threadIdx.x & 31;
  int wave = blockIdx.x * (blockDim.x >> 5) + (threadIdx.x >> 5);
  int nWaves = gridDim.x * (blockDim.x >> 5);
  int nTiles = (n + 15) >> 4;
  int col = lane & 15;
  int hh  = (lane & 16) >> 1;

  v16h Ba[4], Bb[4];
#pragma unroll
  for (int nt = 0; nt < 4; ++nt) { Ba[nt] = load_B(W0, 0, nt * 16, lane); Bb[nt] = load_B(W0, 32, nt * 16, lane); }
  float b0l[4], w1l[4];
#pragma unroll
  for (int nt = 0; nt < 4; ++nt) { b0l[nt] = b0[nt * 16 + col]; w1l[nt] = w1[nt * 16 + col]; }
  float bias = b1 ? b1[0] : 0.f;

  for (int t = wave; t < nTiles; t += nWaves) {
    int row  = t * 16 + col;
    int rowc = row < n ? row : n - 1;
    bool full = (t * 16 + 16 <= n);
    const float* xr = X + (size_t)rowc * 64;
    v16h A0 = load_A_f32(xr, 0, lane);
    v16h A1 = load_A_f32(xr, 32, lane);
    float part[8];
#pragma unroll
    for (int r = 0; r < 8; ++r) part[r] = 0.f;
#pragma unroll
    for (int nt = 0; nt < 4; ++nt) {
      v8f c = zero8();
      c = wmma16(A0, Ba[nt], c);
      c = wmma16(A1, Bb[nt], c);
#pragma unroll
      for (int r = 0; r < 8; ++r) {
        float h = c[r] + b0l[nt];
        h = h > 0.f ? h : 0.f;
        part[r] += h * w1l[nt];
      }
    }
    float* ob = out + t * 16 + hh;
#pragma unroll
    for (int r = 0; r < 8; ++r) {
      float p = part[r];
      p += __shfl_xor(p, 1, 16);
      p += __shfl_xor(p, 2, 16);
      p += __shfl_xor(p, 4, 16);
      p += __shfl_xor(p, 8, 16);
      bool wr = full ? ((lane & 15) == 0) : ((lane & 15) == 0 && (t * 16 + r + hh) < n);
      if (wr) ob[r] = p + bias;
    }
  }
}

extern "C" void kernel_launch(void* const* d_in, const int* in_sizes, int n_in,
                              void* d_out, int out_size, void* d_ws, size_t ws_size,
                              hipStream_t stream) {
  (void)in_sizes; (void)n_in; (void)out_size; (void)ws_size;
  const float* cf = (const float*)d_in[0];
  const int*   ei = (const int*)d_in[1];
  const float* ef = (const float*)d_in[2];
  const float* vf = (const float*)d_in[3];
  const float* cw0 = (const float*)d_in[4],  *cb0 = (const float*)d_in[5];
  const float* cw1 = (const float*)d_in[6],  *cb1 = (const float*)d_in[7];
  const float* vw0 = (const float*)d_in[8],  *vb0 = (const float*)d_in[9];
  const float* vw1 = (const float*)d_in[10], *vb1 = (const float*)d_in[11];
  const float* vc_lw = (const float*)d_in[12], *vc_lb = (const float*)d_in[13];
  const float* vc_ew = (const float*)d_in[14], *vc_rw = (const float*)d_in[15];
  const float* vc_fw = (const float*)d_in[16], *vc_fb = (const float*)d_in[17];
  const float* vc_ow0 = (const float*)d_in[18], *vc_ob0 = (const float*)d_in[19];
  const float* vc_ow1 = (const float*)d_in[20], *vc_ob1 = (const float*)d_in[21];
  const float* cv_lw = (const float*)d_in[22], *cv_lb = (const float*)d_in[23];
  const float* cv_ew = (const float*)d_in[24], *cv_rw = (const float*)d_in[25];
  const float* cv_fw = (const float*)d_in[26], *cv_fb = (const float*)d_in[27];
  const float* cv_ow0 = (const float*)d_in[28], *cv_ob0 = (const float*)d_in[29];
  const float* cv_ow1 = (const float*)d_in[30], *cv_ob1 = (const float*)d_in[31];
  const float* pw0 = (const float*)d_in[32], *pb0 = (const float*)d_in[33];
  const float* pw1 = (const float*)d_in[34];
  const float* vhw0 = (const float*)d_in[35], *vhb0 = (const float*)d_in[36];
  const float* vhw1 = (const float*)d_in[37], *vhb1 = (const float*)d_in[38];

  const int* cons_idx = ei;
  const int* var_idx  = ei + NEDGE;

  char* p = (char*)d_ws;
  auto take = [&](size_t bytes) { void* r = (void*)p; p += (bytes + 255) & ~(size_t)255; return r; };
  _Float16* WH   = (_Float16*)take(256 * 1024);
  float* c_emb   = (float*)take((size_t)NCONS * 64 * 4);
  float* v_emb   = (float*)take((size_t)NVARS * 64 * 4);
  float* agg     = (float*)take((size_t)NVARS * 64 * 4);
  _Float16* Rh   = (_Float16*)take((size_t)NVARS * 64 * 2);
  _Float16* Lh   = (_Float16*)take((size_t)NVARS * 64 * 2);
  float* c2      = (float*)take((size_t)NCONS * 64 * 4);
  float* v2      = (float*)take((size_t)NVARS * 64 * 4);

  _Float16* cw0h = WH + 0;      _Float16* cw1h = WH + 2048;
  _Float16* vw0h = WH + 6144;   _Float16* vw1h = WH + 8192;
  _Float16* vc_lwh = WH + 12288, *vc_rwh = WH + 16384, *vc_fwh = WH + 20480;
  _Float16* vc_ewh = WH + 24576, *vc_ow0h = WH + 24640, *vc_ow1h = WH + 32832;
  _Float16* cv_lwh = WH + 36928, *cv_rwh = WH + 41024, *cv_fwh = WH + 45120;
  _Float16* cv_ewh = WH + 49216, *cv_ow0h = WH + 49280, *cv_ow1h = WH + 57472;
  _Float16* pw0h = WH + 61568,  *vhw0h = WH + 65664;

  auto CVT = [&](const float* s, _Float16* d, int n) {
    k_cvt<<<(n + 255) / 256, 256, 0, stream>>>(s, d, n);
  };
  k_cvt_pad<<<8, 256, 0, stream>>>(cw0, cw0h, 5);
  CVT(cw1, cw1h, 4096);
  k_cvt_pad<<<8, 256, 0, stream>>>(vw0, vw0h, 19);
  CVT(vw1, vw1h, 4096);
  CVT(vc_lw, vc_lwh, 4096); CVT(vc_rw, vc_rwh, 4096); CVT(vc_fw, vc_fwh, 4096);
  CVT(vc_ew, vc_ewh, 64);   CVT(vc_ow0, vc_ow0h, 8192); CVT(vc_ow1, vc_ow1h, 4096);
  CVT(cv_lw, cv_lwh, 4096); CVT(cv_rw, cv_rwh, 4096); CVT(cv_fw, cv_fwh, 4096);
  CVT(cv_ew, cv_ewh, 64);   CVT(cv_ow0, cv_ow0h, 8192); CVT(cv_ow1, cv_ow1h, 4096);
  CVT(pw0, pw0h, 4096);     CVT(vhw0, vhw0h, 4096);

  dim3 blk(256);
  auto grid = [](int tiles) { int w = tiles < 8192 ? tiles : 8192; return dim3((unsigned)((w + 7) / 8)); };
  int tc = (NCONS + 15) / 16, tv = (NVARS + 15) / 16, te = (NEDGE + 15) / 16;

  // node embeddings
  k_embed<5><<<grid(tc), blk, 0, stream>>>(cf, cw0h, cb0, cw1h, cb1, c_emb, NCONS);
  k_embed<19><<<grid(tv), blk, 0, stream>>>(vf, vw0h, vb0, vw1h, vb1, v_emb, NVARS);

  // conv1: v -> c  (right = c_emb, left = v_emb, dst = cons_idx, src = var_idx)
  k_node_lin<<<grid(tc), blk, 0, stream>>>(c_emb, vc_lwh, vc_lb, Rh, NCONS);
  k_node_lin<<<grid(tv), blk, 0, stream>>>(v_emb, vc_rwh, nullptr, Lh, NVARS);
  hipMemsetAsync(agg, 0, (size_t)NCONS * 64 * 4, stream);
  k_edge<<<grid(te), blk, 0, stream>>>(cons_idx, var_idx, ef, Rh, Lh, vc_ewh, vc_fwh, vc_fb, agg, NEDGE);
  k_update<<<grid(tc), blk, 0, stream>>>(agg, c_emb, vc_ow0h, vc_ob0, vc_ow1h, vc_ob1, c2, NCONS);

  // conv2: c -> v  (right = v_emb, left = c2, dst = var_idx, src = cons_idx)
  k_node_lin<<<grid(tv), blk, 0, stream>>>(v_emb, cv_lwh, cv_lb, Rh, NVARS);
  k_node_lin<<<grid(tc), blk, 0, stream>>>(c2, cv_rwh, nullptr, Lh, NCONS);
  hipMemsetAsync(agg, 0, (size_t)NVARS * 64 * 4, stream);
  k_edge<<<grid(te), blk, 0, stream>>>(var_idx, cons_idx, ef, Rh, Lh, cv_ewh, cv_fwh, cv_fb, agg, NEDGE);
  k_update<<<grid(tv), blk, 0, stream>>>(agg, v_emb, cv_ow0h, cv_ob0, cv_ow1h, cv_ob1, v2, NVARS);

  // heads: d_out = [value (NVARS), policy (NVARS)]
  float* out = (float*)d_out;
  k_head<<<grid(tv), blk, 0, stream>>>(v2, vhw0h, vhb0, vhw1, vhb1, out, NVARS);
  k_head<<<grid(tv), blk, 0, stream>>>(v2, pw0h, pb0, pw1, nullptr, out + NVARS, NVARS);
}
